// LeapUncertainty_84567906058936
// MI455X (gfx1250) — matrix operand order repaired and checked by reading, not compiled, not run on hardware
//
#include <hip/hip_runtime.h>

#define TIN   30
#define TOUT  30
#define HDIM  128
#define PITCH 136   // bf16 elements per LDS row (128 + 8 pad -> 4-bank shift per row)
#define MT    4     // M tiles per block (rows = 16*MT = 64)
#define ROWS  (16 * MT)

typedef __attribute__((ext_vector_type(16))) __bf16 v16bf;
typedef __attribute__((ext_vector_type(8)))  float  v8f;

__device__ __forceinline__ v8f splat8(float v){
  v8f r;
#pragma unroll
  for (int i = 0; i < 8; ++i) r[i] = v;
  return r;
}

__device__ __forceinline__ float sig_(float x){ return 1.0f / (1.0f + __expf(-x)); }

#if defined(__has_builtin)
#if __has_builtin(__builtin_amdgcn_tanhf)
#define HAVE_TANH_BUILTIN 1
#endif
#endif
#ifdef HAVE_TANH_BUILTIN
__device__ __forceinline__ float tanh_(float x){ return __builtin_amdgcn_tanhf(x); }
#else
__device__ __forceinline__ float tanh_(float x){ return 2.0f / (1.0f + __expf(-2.0f * x)) - 1.0f; }
#endif

__device__ __forceinline__ v8f wmma_bf16(v16bf a, v16bf b, v8f c){
  return __builtin_amdgcn_wmma_f32_16x16x32_bf16(false, a, false, b, (short)0, c, false, false);
}

// Fragment with K=0..2 valid, rest zero (valid for both the A-side input
// fragment and the B-side Wih0 fragment: lanes >= 16 carry zeros).
__device__ __forceinline__ v16bf mk3(float a0, float a1, float a2){
  v16bf f;
#pragma unroll
  for (int i = 0; i < 16; ++i) f[i] = (__bf16)0.0f;
  f[0] = (__bf16)a0; f[1] = (__bf16)a1; f[2] = (__bf16)a2;
  return f;
}

// A fragment (16x32 bf16) from LDS buffer laid out [m][k], pitch PITCH.
__device__ __forceinline__ v16bf ldA(const __bf16* Hb, int mt, int kt, int lane){
  const __bf16* p = Hb + (mt * 16 + (lane & 15)) * PITCH + kt * 32 + (lane >> 4) * 16;
  union { uint4 q[2]; v16bf f; } u;
  u.q[0] = *(const uint4*)(p);
  u.q[1] = *(const uint4*)(p + 8);
  return u.f;
}

// B fragment (32x16 bf16) from LDS weight matrix laid out [n][k], pitch PITCH.
__device__ __forceinline__ v16bf ldB(const __bf16* W, int nb, int kt, int lane){
  const __bf16* p = W + (nb + (lane & 15)) * PITCH + kt * 32 + (lane >> 4) * 16;
  union { uint4 q[2]; v16bf f; } u;
  u.q[0] = *(const uint4*)(p);
  u.q[1] = *(const uint4*)(p + 8);
  return u.f;
}

// Build a B fragment directly from a global f32 weight matrix W[512][128] (row-major).
__device__ __forceinline__ v16bf mkBW(const float* W, int nb, int kt, int lane){
  const float* p = W + (long)(nb + (lane & 15)) * HDIM + kt * 32 + (lane >> 4) * 16;
  v16bf f;
#pragma unroll
  for (int i = 0; i < 16; ++i) f[i] = (__bf16)p[i];
  return f;
}

// Convert one 512x128 f32 matrix (row-major) to bf16 LDS with pitch PITCH.
__device__ __forceinline__ void fill512(__bf16* dst, const float* src){
  for (int i = threadIdx.x; i < 512 * 32; i += 256){
    int r = i >> 5;
    int c = (i & 31) << 2;
    const float* p = src + (long)r * HDIM + c;
    __bf16* q = dst + r * PITCH + c;
    q[0] = (__bf16)p[0]; q[1] = (__bf16)p[1];
    q[2] = (__bf16)p[2]; q[3] = (__bf16)p[3];
  }
}

__global__ void __launch_bounds__(256, 2)
seq2seq_lstm(const float* __restrict__ xg,
             const float* __restrict__ eWih0, const float* __restrict__ eWhh0, const float* __restrict__ eb0,
             const float* __restrict__ eWih1, const float* __restrict__ eWhh1, const float* __restrict__ eb1,
             const float* __restrict__ dWih0, const float* __restrict__ dWhh0, const float* __restrict__ db0,
             const float* __restrict__ dWih1, const float* __restrict__ dWhh1, const float* __restrict__ db1,
             const float* __restrict__ fcW, const float* __restrict__ fcb,
             float* __restrict__ outp, int Bn)
{
  __shared__ __align__(16) __bf16 sW0[512 * PITCH];   // Whh0 (bf16, [n][k])
  __shared__ __align__(16) __bf16 sW1[512 * PITCH];   // Whh1 (bf16, [n][k])
  __shared__ __align__(16) __bf16 sFC[16 * PITCH];    // FC weights (rows 6..15 zero)
  __shared__ __align__(16) __bf16 sH0[ROWS * PITCH];  // h0 state, [m][k]
  __shared__ __align__(16) __bf16 sH1[ROWS * PITCH];  // h1 state, [m][k]
  __shared__ float sX[ROWS * 4];                      // decoder feedback input (mu)

  const int lane = threadIdx.x & 31;
  const int wave = threadIdx.x >> 5;
  const int ln   = lane & 15;
  const bool hi  = lane >= 16;
  const int cb   = wave << 4;                 // hidden-column base owned by this wave
  const long rowbase = (long)blockIdx.x * ROWS;
  float* outmu = outp;
  float* outlv = outp + (long)Bn * TOUT * 3;

  for (int i = threadIdx.x; i < ROWS * PITCH; i += 256){
    sH0[i] = (__bf16)0.0f;
    sH1[i] = (__bf16)0.0f;
  }

  v8f c0[MT], c1[MT];
#pragma unroll
  for (int mt = 0; mt < MT; ++mt){ c0[mt] = splat8(0.0f); c1[mt] = splat8(0.0f); }

  for (int phase = 0; phase < 2; ++phase){
    const float* Wih0 = phase ? dWih0 : eWih0;
    const float* Whh0 = phase ? dWhh0 : eWhh0;
    const float* b0   = phase ? db0   : eb0;
    const float* Wih1 = phase ? dWih1 : eWih1;
    const float* Whh1 = phase ? dWhh1 : eWhh1;
    const float* b1   = phase ? db1   : eb1;

    __syncthreads();   // previous phase done with sW* (also covers sH zero-fill)
    fill512(sW0, Whh0);
    fill512(sW1, Whh1);
    if (phase){
      for (int i = threadIdx.x; i < 16 * HDIM; i += 256){
        int r = i >> 7, k = i & 127;
        sFC[r * PITCH + k] = (__bf16)(r < 6 ? fcW[r * HDIM + k] : 0.0f);
      }
    }

    // Wih1 N-slice stationary in registers: 4 gates x 4 K-tiles = 16 fragments.
    v16bf w1f[4][4];
#pragma unroll
    for (int gt = 0; gt < 4; ++gt)
#pragma unroll
      for (int kt = 0; kt < 4; ++kt)
        w1f[gt][kt] = mkBW(Wih1, gt * HDIM + cb, kt, lane);

    // Wih0 kept as 12 packed scalars (zero on lanes >= 16); fragments rebuilt per use.
    float w0x[4], w0y[4], w0z[4];
#pragma unroll
    for (int gt = 0; gt < 4; ++gt){
      const float* p = Wih0 + (gt * HDIM + cb + ln) * 3;  // lanes >= 16 duplicate 0..15
      float a0 = p[0], a1 = p[1], a2 = p[2];
      w0x[gt] = hi ? 0.0f : a0;
      w0y[gt] = hi ? 0.0f : a1;
      w0z[gt] = hi ? 0.0f : a2;
    }

    float bia0[4], bia1[4];
#pragma unroll
    for (int gt = 0; gt < 4; ++gt){
      bia0[gt] = b0[gt * HDIM + cb + ln];
      bia1[gt] = b1[gt * HDIM + cb + ln];
    }
    float bfc = 0.0f;
    if (phase){
      float v = fcb[ln < 6 ? ln : 0];
      bfc = (ln < 6) ? v : 0.0f;
    }

    __syncthreads();   // weights visible

    for (int t = 0; t < 30; ++t){
      // ---------------- layer 0 ----------------
      v8f acc[MT][4];
#pragma unroll
      for (int mt = 0; mt < MT; ++mt)
#pragma unroll
        for (int gt = 0; gt < 4; ++gt) acc[mt][gt] = splat8(bia0[gt]);

      // input contribution (K=3, zero-padded fragments); branchless per-lane:
      // lanes >= 16 duplicate the loads of lanes 0..15 and are masked to zero.
#pragma unroll
      for (int mt = 0; mt < MT; ++mt){
        float x0, x1, x2;
        if (phase == 0 || t == 0){                 // uniform scalar branch
          long row = rowbase + mt * 16 + ln;
          const float* p = xg + row * (TIN * 3) + (phase ? (TIN - 1) * 3 : t * 3);
          x0 = p[0]; x1 = p[1]; x2 = p[2];
        } else {
          int m = mt * 16 + ln;
          x0 = sX[m * 4 + 0]; x1 = sX[m * 4 + 1]; x2 = sX[m * 4 + 2];
        }
        x0 = hi ? 0.0f : x0;
        x1 = hi ? 0.0f : x1;
        x2 = hi ? 0.0f : x2;
        v16bf xa = mk3(x0, x1, x2);
#pragma unroll
        for (int gt = 0; gt < 4; ++gt)
          acc[mt][gt] = wmma_bf16(xa, mk3(w0x[gt], w0y[gt], w0z[gt]), acc[mt][gt]);
      }

      // recurrent part: B fragments shared by all M tiles
#pragma unroll
      for (int kt = 0; kt < 4; ++kt){
        v16bf a[MT];
#pragma unroll
        for (int mt = 0; mt < MT; ++mt) a[mt] = ldA(sH0, mt, kt, lane);
#pragma unroll
        for (int gt = 0; gt < 4; ++gt){
          v16bf b = ldB(sW0, gt * HDIM + cb, kt, lane);
#pragma unroll
          for (int mt = 0; mt < MT; ++mt)
            acc[mt][gt] = wmma_bf16(a[mt], b, acc[mt][gt]);
        }
      }

      v8f hn0[MT];
#pragma unroll
      for (int mt = 0; mt < MT; ++mt){
        v8f cc = c0[mt], hh;
#pragma unroll
        for (int e = 0; e < 8; ++e){
          float iv = sig_(acc[mt][0][e]);
          float fv = sig_(acc[mt][1][e]);
          float gv = tanh_(acc[mt][2][e]);
          float ov = sig_(acc[mt][3][e]);
          float cv = fv * cc[e] + iv * gv;
          cc[e] = cv;
          hh[e] = ov * tanh_(cv);
        }
        c0[mt] = cc; hn0[mt] = hh;
      }
      __syncthreads();   // all waves finished reading old sH0
#pragma unroll
      for (int mt = 0; mt < MT; ++mt)
#pragma unroll
        for (int e = 0; e < 8; ++e)
          sH0[(mt * 16 + (lane >> 4) * 8 + e) * PITCH + cb + ln] = (__bf16)hn0[mt][e];
      __syncthreads();   // new sH0 visible

      // ---------------- layer 1 ----------------
#pragma unroll
      for (int mt = 0; mt < MT; ++mt)
#pragma unroll
        for (int gt = 0; gt < 4; ++gt) acc[mt][gt] = splat8(bia1[gt]);

      // h0 @ Wih1^T : Wih1 fragments from registers
#pragma unroll
      for (int kt = 0; kt < 4; ++kt){
        v16bf a[MT];
#pragma unroll
        for (int mt = 0; mt < MT; ++mt) a[mt] = ldA(sH0, mt, kt, lane);
#pragma unroll
        for (int gt = 0; gt < 4; ++gt)
#pragma unroll
          for (int mt = 0; mt < MT; ++mt)
            acc[mt][gt] = wmma_bf16(a[mt], w1f[gt][kt], acc[mt][gt]);
      }
      // h1 @ Whh1^T : B fragments from LDS, shared by all M tiles
#pragma unroll
      for (int kt = 0; kt < 4; ++kt){
        v16bf a[MT];
#pragma unroll
        for (int mt = 0; mt < MT; ++mt) a[mt] = ldA(sH1, mt, kt, lane);
#pragma unroll
        for (int gt = 0; gt < 4; ++gt){
          v16bf b = ldB(sW1, gt * HDIM + cb, kt, lane);
#pragma unroll
          for (int mt = 0; mt < MT; ++mt)
            acc[mt][gt] = wmma_bf16(a[mt], b, acc[mt][gt]);
        }
      }

      v8f hn1[MT];
#pragma unroll
      for (int mt = 0; mt < MT; ++mt){
        v8f cc = c1[mt], hh;
#pragma unroll
        for (int e = 0; e < 8; ++e){
          float iv = sig_(acc[mt][0][e]);
          float fv = sig_(acc[mt][1][e]);
          float gv = tanh_(acc[mt][2][e]);
          float ov = sig_(acc[mt][3][e]);
          float cv = fv * cc[e] + iv * gv;
          cc[e] = cv;
          hh[e] = ov * tanh_(cv);
        }
        c1[mt] = cc; hn1[mt] = hh;
      }
      __syncthreads();   // all waves finished reading old sH1
#pragma unroll
      for (int mt = 0; mt < MT; ++mt)
#pragma unroll
        for (int e = 0; e < 8; ++e)
          sH1[(mt * 16 + (lane >> 4) * 8 + e) * PITCH + cb + ln] = (__bf16)hn1[mt][e];
      __syncthreads();   // new sH1 visible

      // ---------------- decoder FC head ----------------
      if (phase){
        if (wave == 0){
          v8f fa[MT];
#pragma unroll
          for (int mt = 0; mt < MT; ++mt) fa[mt] = splat8(bfc);
#pragma unroll
          for (int kt = 0; kt < 4; ++kt){
            v16bf b = ldB(sFC, 0, kt, lane);
#pragma unroll
            for (int mt = 0; mt < MT; ++mt)
              fa[mt] = wmma_bf16(ldA(sH1, mt, kt, lane), b, fa[mt]);
          }
#pragma unroll
          for (int mt = 0; mt < MT; ++mt){
#pragma unroll
            for (int e = 0; e < 8; ++e){
              int m = mt * 16 + (lane >> 4) * 8 + e;
              long row = rowbase + m;
              float v = fa[mt][e];
              if (ln < 3){
                outmu[row * (TOUT * 3) + t * 3 + ln] = v;
                sX[m * 4 + ln] = v;                      // feedback input for t+1
              } else if (ln < 6){
                outlv[row * (TOUT * 3) + t * 3 + (ln - 3)] = v;
              }
            }
          }
        }
        __syncthreads();   // sX visible to all waves
      }
    }
  }
}

extern "C" void kernel_launch(void* const* d_in, const int* in_sizes, int n_in,
                              void* d_out, int out_size, void* d_ws, size_t ws_size,
                              hipStream_t stream)
{
  const float* x     = (const float*)d_in[0];
  const float* eWih0 = (const float*)d_in[1];
  const float* eWhh0 = (const float*)d_in[2];
  const float* eb0   = (const float*)d_in[3];
  const float* eWih1 = (const float*)d_in[4];
  const float* eWhh1 = (const float*)d_in[5];
  const float* eb1   = (const float*)d_in[6];
  const float* dWih0 = (const float*)d_in[7];
  const float* dWhh0 = (const float*)d_in[8];
  const float* db0   = (const float*)d_in[9];
  const float* dWih1 = (const float*)d_in[10];
  const float* dWhh1 = (const float*)d_in[11];
  const float* db1   = (const float*)d_in[12];
  const float* fcW   = (const float*)d_in[13];
  const float* fcb   = (const float*)d_in[14];

  int Bn = in_sizes[0] / (TIN * 3);
  dim3 grid((unsigned)(Bn / ROWS)), block(256);
  hipLaunchKernelGGL(seq2seq_lstm, grid, block, 0, stream,
                     x, eWih0, eWhh0, eb0, eWih1, eWhh1, eb1,
                     dWih0, dWhh0, db0, dWih1, dWhh1, db1,
                     fcW, fcb, (float*)d_out, Bn);
}
